// DreamAttention_4148938408885
// MI455X (gfx1250) — matrix operand order repaired and checked by reading, not compiled, not run on hardware
//
#include <hip/hip_runtime.h>
#include <hip/hip_bf16.h>

// ---------------------------------------------------------------------------
// Problem constants (reference: B=2, S=2048, D=2048, NH=16, HD=128)
// ---------------------------------------------------------------------------
#define B_  2
#define S_  2048
#define D_  2048
#define NH_ 16
#define HD_ 128
#define M_  (B_ * S_)          // 4096 rows for all projection GEMMs

typedef __attribute__((ext_vector_type(16))) __bf16 v16bf;
typedef __attribute__((ext_vector_type(8)))  __bf16 v8bf;
typedef __attribute__((ext_vector_type(8)))  float  v8f;

union V16U { v16bf v; v8bf h[2]; };

// ---------------------------------------------------------------------------
// WMMA operand loaders (CDNA5 16x16x32 bf16 layouts, ISA 7.12.2, wave32)
// A-matrix 16x32: lane m=lane&15; lanes0-15 hold K 0..7 & 16..23,
//                 lanes16-31 hold K 8..15 & 24..31 (two 16B chunks).
// B-matrix 32x16: lane n=lane&15; lanes0-15 hold K 0..7 & 16..23, lanes16-31
//                 K 8..15 & 24..31 (two 16B chunks). Source stored K-major per row n.
// ---------------------------------------------------------------------------
__device__ __forceinline__ v16bf load_a16x32(const __bf16* p, int ld) {
  int lane = threadIdx.x & 31;
  int m    = lane & 15;
  int ko   = (lane >> 4) << 3;            // 0 or 8
  const __bf16* row = p + (size_t)m * ld;
  V16U u;
  u.h[0] = *(const v8bf*)(row + ko);      // K = ko .. ko+7
  u.h[1] = *(const v8bf*)(row + 16 + ko); // K = 16+ko .. 16+ko+7
  return u.v;
}

__device__ __forceinline__ v16bf load_b32x16(const __bf16* p, int ld) {
  int lane = threadIdx.x & 31;
  int n    = lane & 15;
  int ko   = (lane >> 4) << 3;            // 0 or 8
  const __bf16* row = p + (size_t)n * ld;
  V16U u;
  u.h[0] = *(const v8bf*)(row + ko);
  u.h[1] = *(const v8bf*)(row + 16 + ko);
  return u.v;
}

__device__ __forceinline__ v8f wmma_bf16(v16bf a, v16bf b, v8f c) {
  return __builtin_amdgcn_wmma_f32_16x16x32_bf16(false, a, false, b,
                                                 (short)0, c, false, false);
}

// ---------------------------------------------------------------------------
// Tensor Data Mover support (gfx1250).  Guarded: device pass + builtin present.
// ---------------------------------------------------------------------------
#if defined(__HIP_DEVICE_COMPILE__) && defined(__has_builtin)
#if __has_builtin(__builtin_amdgcn_tensor_load_to_lds) && \
    __has_builtin(__builtin_amdgcn_s_wait_tensorcnt)
#define CDNA5_TDM 1
#endif
#endif

#ifdef CDNA5_TDM
typedef __attribute__((ext_vector_type(4))) unsigned int u32x4;
typedef __attribute__((ext_vector_type(8))) int          i32x8;
typedef __attribute__((ext_vector_type(4))) int          i32x4;

// DMA a 128-row x 32-col bf16 tile (row stride K elements) from global
// memory into LDS (rows packed contiguously, 64B apart).
__device__ __forceinline__ void tdm_load_B(__bf16* lds_dst,
                                           const __bf16* gsrc, int K) {
  // Generic LDS address: low 32 bits are the hardware LDS byte address.
  unsigned lds_addr = (unsigned)(unsigned long long)(size_t)lds_dst;
  unsigned long long ga = (unsigned long long)(size_t)gsrc;

  u32x4 g0;
  g0[0] = 1u;                                            // count=1, user D#
  g0[1] = lds_addr;                                      // lds_addr (bytes)
  g0[2] = (unsigned)(ga & 0xFFFFFFFFu);                  // global_addr[31:0]
  g0[3] = (unsigned)((ga >> 32) & 0x01FFFFFFu) | (2u << 30); // [56:32] | type=2

  i32x8 g1;
  g1[0] = (int)(1u << 16);              // workgroup_mask=0, data_size=1 (2B)
  g1[1] = (int)(32u << 16);             // tensor_dim0 = 32 (low 16 @ [31:16])
  g1[2] = (int)(128u << 16);            // dim0 hi=0 | tensor_dim1=128 low16
  g1[3] = (int)(32u << 16);             // dim1 hi=0 | tile_dim0 = 32
  g1[4] = (int)(128u);                  // tile_dim1 = 128, tile_dim2 = 0
  g1[5] = (int)(unsigned)K;             // tensor_dim0_stride low 32 = K
  g1[6] = 0;                            // stride0 hi | stride1 lo (unused)
  g1[7] = 0;

  i32x4 z4 = {0, 0, 0, 0};
#if __clang_major__ >= 23
  i32x8 z8 = {};
  __builtin_amdgcn_tensor_load_to_lds(g0, g1, z4, z4, z8, 0);
#else
  __builtin_amdgcn_tensor_load_to_lds(g0, g1, z4, z4, 0);
#endif
}
#endif  // CDNA5_TDM

// ---------------------------------------------------------------------------
// fp32 -> bf16 cast
// ---------------------------------------------------------------------------
__global__ void cast_f32_bf16(const float* __restrict__ src,
                              __bf16* __restrict__ dst, int n) {
  int i = blockIdx.x * blockDim.x + threadIdx.x;
  if (i < n) dst[i] = (__bf16)src[i];
}

// ---------------------------------------------------------------------------
// C(MxN) = A(MxK) @ W(NxK)^T + bias   (bf16 in, fp32 out)
// Block = 256 threads = 8 waves (4 in M x 2 in N); block tile 128x128;
// wave tile 32x64 (8 f32 accumulators).  B block-tile (128x32 = 8KB) is
// staged in LDS, double-buffered via the Tensor Data Mover (wave 0 issues
// tensor_load_to_lds for chunk k+32 while the block consumes chunk k, with
// s_wait_tensorcnt covering the in-flight buffer).  Fallback: cooperative
// global->LDS staging.
// ---------------------------------------------------------------------------
__global__ __launch_bounds__(256)
void gemm_bf16_wmma(const __bf16* __restrict__ A,
                    const __bf16* __restrict__ W,
                    const float*  __restrict__ bias,   // may be null
                    float* __restrict__ Cout,
                    int M, int N, int K) {
  __shared__ __align__(32) __bf16 Bs[2][128][32];

  int w   = threadIdx.x >> 5;
  int wm  = w >> 1, wn = w & 1;
  int m0  = blockIdx.y * 128 + wm * 32;
  int nb0 = blockIdx.x * 128;
  int n0  = nb0 + wn * 64;

  v8f accLo[4] = {}, accHi[4] = {};
  const __bf16* arow0 = A + (size_t)m0 * K;
  const __bf16* arow1 = A + (size_t)(m0 + 16) * K;

#ifdef CDNA5_TDM
  if (w == 0) tdm_load_B(&Bs[0][0][0], W + (size_t)nb0 * K, K);
#endif

  int buf = 0;
  for (int k = 0; k < K; k += 32) {
#ifdef CDNA5_TDM
    if (w == 0) {
      if (k + 32 < K) {
        tdm_load_B(&Bs[buf ^ 1][0][0], W + (size_t)nb0 * K + (k + 32), K);
        __builtin_amdgcn_s_wait_tensorcnt(1);  // current buffer landed
      } else {
        __builtin_amdgcn_s_wait_tensorcnt(0);
      }
    }
#else
    {   // cooperative staging: 256 threads x 32B = 8KB tile
      int row = threadIdx.x >> 1, half = threadIdx.x & 1;
      *(v16bf*)&Bs[buf][row][half * 16] =
          *(const v16bf*)(W + (size_t)(nb0 + row) * K + k + half * 16);
    }
#endif
    __syncthreads();

    v16bf a0 = load_a16x32(arow0 + k, K);
    v16bf a1 = load_a16x32(arow1 + k, K);
    __builtin_prefetch(arow0 + k + 64, 0, 1);   // global_prefetch_b8
    __builtin_prefetch(arow1 + k + 64, 0, 1);
#pragma unroll
    for (int j = 0; j < 4; ++j) {
      v16bf b = load_b32x16(&Bs[buf][wn * 64 + j * 16][0], 32);
      accLo[j] = wmma_bf16(a0, b, accLo[j]);
      accHi[j] = wmma_bf16(a1, b, accHi[j]);
    }
    __syncthreads();
    buf ^= 1;
  }

  int lane = threadIdx.x & 31;
  int nn   = lane & 15;
  int mb   = (lane >> 4) << 3;
#pragma unroll
  for (int j = 0; j < 4; ++j) {
    float bv = bias ? bias[n0 + j * 16 + nn] : 0.f;
#pragma unroll
    for (int r = 0; r < 8; ++r) {
      Cout[(size_t)(m0 +  0 + mb + r) * N + n0 + j * 16 + nn] = accLo[j][r] + bv;
      Cout[(size_t)(m0 + 16 + mb + r) * N + n0 + j * 16 + nn] = accHi[j][r] + bv;
    }
  }
}

// ---------------------------------------------------------------------------
// RoPE + repack: X fp32 (B,S,D) -> Y bf16 (B,NH,S,HD) with rotary embedding.
// angle[d] = pos * theta^(-(d%64)/64);  q' = q*cos + rotate_half(q)*sin
// ---------------------------------------------------------------------------
__global__ void rope_pack(const float* __restrict__ X,
                          const int*   __restrict__ pos_ids,
                          __bf16* __restrict__ Y) {
  int i = blockIdx.x * blockDim.x + threadIdx.x;
  if (i >= B_ * S_ * D_) return;
  int d = i & (HD_ - 1);
  int h = (i / HD_) & (NH_ - 1);
  int s = (i / D_) & (S_ - 1);
  int b = i / (S_ * D_);

  int   pos = pos_ids[b * S_ + s];
  int   d64 = d & 63;
  // ln(10000)/64 = 0.14391156831...
  float ang = (float)pos * __expf(-(float)d64 * 0.14391156831f);
  float sn, cs;
  __sincosf(ang, &sn, &cs);
  float x  = X[i];
  float xr = (d < 64) ? -X[i + 64] : X[i - 64];
  float y  = x * cs + xr * sn;
  Y[((size_t)(b * NH_ + h) * S_ + s) * HD_ + d] = (__bf16)y;
}

// ---------------------------------------------------------------------------
// V repack transposed: X fp32 (B,S,D) -> Vt bf16 (B,NH,HD,S)
// ---------------------------------------------------------------------------
__global__ void vpack_t(const float* __restrict__ X, __bf16* __restrict__ Vt) {
  int i = blockIdx.x * blockDim.x + threadIdx.x;
  if (i >= B_ * S_ * D_) return;
  int d = i & (HD_ - 1);
  int h = (i / HD_) & (NH_ - 1);
  int s = (i / D_) & (S_ - 1);
  int b = i / (S_ * D_);
  Vt[((size_t)(b * NH_ + h) * HD_ + d) * S_ + s] = (__bf16)X[i];
}

// ---------------------------------------------------------------------------
// Attention (no softmax in reference!):  O = (Q K^T) V / sqrt(HD)
// Q,K: (B*NH, S, HD) bf16;  Vt: (B*NH, HD, S) bf16;  Att out: (B,S,D) bf16.
// Grid (S/128, B*NH); 8 waves/block; each wave owns 16 query rows.
// Per 32-key iteration: 8 WMMAs for QK^T, LDS round-trip to re-layout the
// score tile (C/D layout -> A layout), then 8 WMMAs for P@V.
// ---------------------------------------------------------------------------
__global__ __launch_bounds__(256)
void attn_wmma(const __bf16* __restrict__ Qb,
               const __bf16* __restrict__ Kb,
               const __bf16* __restrict__ Vt,
               __bf16* __restrict__ Att) {
  __shared__ __align__(32) __bf16 P[8][16][32];

  int w    = threadIdx.x >> 5;
  int lane = threadIdx.x & 31;
  int bh   = blockIdx.y;
  int b    = bh / NH_, h = bh % NH_;
  int q0   = blockIdx.x * 128 + w * 16;

  const __bf16* Q = Qb + (size_t)bh * S_ * HD_;
  const __bf16* K = Kb + (size_t)bh * S_ * HD_;
  const __bf16* V = Vt + (size_t)bh * HD_ * S_;

  v16bf qa[4];
#pragma unroll
  for (int c = 0; c < 4; ++c)
    qa[c] = load_a16x32(Q + (size_t)q0 * HD_ + c * 32, HD_);

  v8f o[8] = {};
  int nn = lane & 15;
  int mb = (lane >> 4) << 3;

  for (int kk = 0; kk < S_; kk += 32) {
    // ---- scores for 32 keys
    v8f s0 = {}, s1 = {};
#pragma unroll
    for (int c = 0; c < 4; ++c) {
      v16bf bk0 = load_b32x16(K + (size_t)(kk +  0) * HD_ + c * 32, HD_);
      v16bf bk1 = load_b32x16(K + (size_t)(kk + 16) * HD_ + c * 32, HD_);
      s0 = wmma_bf16(qa[c], bk0, s0);
      s1 = wmma_bf16(qa[c], bk1, s1);
    }
    // ---- re-layout 16x32 score tile via LDS (C/D layout -> A-operand layout)
#pragma unroll
    for (int r = 0; r < 8; ++r) {
      P[w][mb + r][nn]      = (__bf16)s0[r];
      P[w][mb + r][16 + nn] = (__bf16)s1[r];
    }
    __syncthreads();
    v16bf pa = load_a16x32(&P[w][0][0], 32);
    // ---- O(16x128) += P(16x32) @ V(32x128)
#pragma unroll
    for (int j = 0; j < 8; ++j) {
      v16bf bv = load_b32x16(V + (size_t)(j * 16) * S_ + kk, S_);
      o[j] = wmma_bf16(pa, bv, o[j]);
    }
    __syncthreads();
  }

  const float scale = 0.08838834764831845f;   // 1/sqrt(128)
#pragma unroll
  for (int j = 0; j < 8; ++j)
#pragma unroll
    for (int r = 0; r < 8; ++r) {
      int m = q0 + mb + r;
      Att[((size_t)(b * S_ + m)) * D_ + h * HD_ + j * 16 + nn] =
          (__bf16)(o[j][r] * scale);
    }
}

// ---------------------------------------------------------------------------
// Launch
// ---------------------------------------------------------------------------
extern "C" void kernel_launch(void* const* d_in, const int* in_sizes, int n_in,
                              void* d_out, int out_size, void* d_ws, size_t ws_size,
                              hipStream_t stream) {
  (void)in_sizes; (void)n_in; (void)out_size; (void)ws_size;

  const float* hs  = (const float*)d_in[0];
  const int*   pid = (const int*)  d_in[1];
  const float* Wq  = (const float*)d_in[2];
  const float* bq  = (const float*)d_in[3];
  const float* Wk  = (const float*)d_in[4];
  const float* bk  = (const float*)d_in[5];
  const float* Wv  = (const float*)d_in[6];
  const float* bv  = (const float*)d_in[7];
  const float* Wo  = (const float*)d_in[8];
  float* out = (float*)d_out;

  char*  ws  = (char*)d_ws;
  size_t off = 0;
  auto alloc = [&](size_t bytes) -> char* {
    char* p = ws + off;
    off += (bytes + 255) & ~(size_t)255;
    return p;
  };

  const size_t nMD = (size_t)M_ * D_;    // 8,388,608
  const size_t nDD = (size_t)D_ * D_;    // 4,194,304

  __bf16* hsb = (__bf16*)alloc(nMD * 2);
  __bf16* wqb = (__bf16*)alloc(nDD * 2);
  __bf16* wkb = (__bf16*)alloc(nDD * 2);
  __bf16* wvb = (__bf16*)alloc(nDD * 2);
  __bf16* wob = (__bf16*)alloc(nDD * 2);
  float*  Qf  = (float*) alloc(nMD * 4);
  float*  Kf  = (float*) alloc(nMD * 4);
  float*  Vf  = (float*) alloc(nMD * 4);
  __bf16* Qr  = (__bf16*)alloc(nMD * 2);
  __bf16* Kr  = (__bf16*)alloc(nMD * 2);
  __bf16* Vt  = (__bf16*)alloc(nMD * 2);
  __bf16* At  = (__bf16*)alloc(nMD * 2);

  const int T = 256;
  cast_f32_bf16<<<(int)((nMD + T - 1) / T), T, 0, stream>>>(hs, hsb, (int)nMD);
  cast_f32_bf16<<<(int)((nDD + T - 1) / T), T, 0, stream>>>(Wq, wqb, (int)nDD);
  cast_f32_bf16<<<(int)((nDD + T - 1) / T), T, 0, stream>>>(Wk, wkb, (int)nDD);
  cast_f32_bf16<<<(int)((nDD + T - 1) / T), T, 0, stream>>>(Wv, wvb, (int)nDD);
  cast_f32_bf16<<<(int)((nDD + T - 1) / T), T, 0, stream>>>(Wo, wob, (int)nDD);

  dim3 gg(D_ / 128, M_ / 128);   // 128x128 block tiles
  gemm_bf16_wmma<<<gg, T, 0, stream>>>(hsb, wqb, bq, Qf, M_, D_, D_);
  gemm_bf16_wmma<<<gg, T, 0, stream>>>(hsb, wkb, bk, Kf, M_, D_, D_);
  gemm_bf16_wmma<<<gg, T, 0, stream>>>(hsb, wvb, bv, Vf, M_, D_, D_);

  rope_pack<<<(int)((nMD + T - 1) / T), T, 0, stream>>>(Qf, pid, Qr);
  rope_pack<<<(int)((nMD + T - 1) / T), T, 0, stream>>>(Kf, pid, Kr);
  vpack_t  <<<(int)((nMD + T - 1) / T), T, 0, stream>>>(Vf, Vt);

  dim3 ga(S_ / 128, B_ * NH_);
  attn_wmma<<<ga, T, 0, stream>>>(Qr, Kr, Vt, At);

  gemm_bf16_wmma<<<gg, T, 0, stream>>>(At, wob, nullptr, out, M_, D_, D_);
}
